// GCN_3762391351712
// MI455X (gfx1250) — matrix operand order repaired and checked by reading, not compiled, not run on hardware
//
#include <hip/hip_runtime.h>
#include <hip/hip_bf16.h>

typedef __bf16 bf16_t;
typedef bf16_t bf16x8  __attribute__((ext_vector_type(8)));
typedef bf16_t bf16x16 __attribute__((ext_vector_type(16)));
typedef float  v8f     __attribute__((ext_vector_type(8)));
typedef int    v4i     __attribute__((ext_vector_type(4)));

#define N_NODES 50000
#define N_EDGES 800000
#define NFEAT   128
#define NHID    256
#define NOUT    128
#define MTILES  (N_NODES / 16)   // 3125 exactly

// CDNA5 async global->LDS path, guarded so host pass / older toolchains fall back
#if defined(__has_builtin)
#  if __has_builtin(__builtin_amdgcn_global_load_async_to_lds_b128) && \
      __has_builtin(__builtin_amdgcn_s_wait_asynccnt)
#    define USE_ASYNC_LDS 1
#  endif
#endif

// ---------------------------------------------------------------------------
// fp32 -> bf16 straight conversion (for x)
// ---------------------------------------------------------------------------
__global__ __launch_bounds__(256) void convert_bf16_kernel(
    const float* __restrict__ in, bf16_t* __restrict__ out, int total)
{
    for (int i = blockIdx.x * blockDim.x + threadIdx.x; i < total;
         i += gridDim.x * blockDim.x)
        out[i] = (bf16_t)in[i];
}

// W[K][N] fp32 (row-major) -> Wt[N][K] bf16 (row-major)
__global__ __launch_bounds__(256) void transpose_bf16_kernel(
    const float* __restrict__ W, bf16_t* __restrict__ Wt, int K, int N)
{
    int total = K * N;
    for (int i = blockIdx.x * blockDim.x + threadIdx.x; i < total;
         i += gridDim.x * blockDim.x) {
        int n = i / K, k = i % K;
        Wt[i] = (bf16_t)W[(size_t)k * N + n];
    }
}

// agg[n][c] = bias[c]
__global__ __launch_bounds__(256) void init_bias_kernel(
    float* __restrict__ agg, const float* __restrict__ bias, int total, int C)
{
    for (int i = blockIdx.x * blockDim.x + threadIdx.x; i < total;
         i += gridDim.x * blockDim.x)
        agg[i] = bias[i & (C - 1)];   // C is a power of two (256 / 128)
}

// ---------------------------------------------------------------------------
// BF16 WMMA GEMM:  C[M,N] (fp32) = A[M,K] (bf16, row-major) @ Bt[N,K]^T
// Block = 256 threads = 8 waves; each wave owns one 16x16 C tile,
// block shares a 16-column B panel staged in LDS (async DMA when available).
// ---------------------------------------------------------------------------
template<int K>
__global__ __launch_bounds__(256) void gemm_bf16_wmma(
    const bf16_t* __restrict__ A, const bf16_t* __restrict__ Bt,
    float* __restrict__ C, int N)
{
    __shared__ __align__(16) bf16_t sB[16 * K];

    const int nBase = blockIdx.y * 16;

    // Stage 16 consecutive rows of Bt (contiguous 16*K bf16) into LDS.
#if defined(USE_ASYNC_LDS)
    {
        // Async global->LDS DMA: 16B per lane per op, no VGPR round-trip,
        // tracked by ASYNCcnt.
        typedef __attribute__((address_space(1))) v4i* gv4i_p;
        typedef __attribute__((address_space(3))) v4i* lv4i_p;
        char* gpanel = (char*)(Bt + (size_t)nBase * K);
        char* lpanel = (char*)sB;
        for (int i = threadIdx.x; i < (16 * K) / 8; i += blockDim.x)
            __builtin_amdgcn_global_load_async_to_lds_b128(
                (gv4i_p)(gpanel + (size_t)i * 16),
                (lv4i_p)(lpanel + (size_t)i * 16),
                0, 0);
        __builtin_amdgcn_s_wait_asynccnt(0);
    }
#else
    {
        const bf16x8* src = (const bf16x8*)(Bt + (size_t)nBase * K);
        bf16x8* dstv = (bf16x8*)sB;
        for (int i = threadIdx.x; i < (16 * K) / 8; i += blockDim.x)
            dstv[i] = src[i];
    }
#endif
    __syncthreads();

    const int wave = threadIdx.x >> 5;
    const int lane = threadIdx.x & 31;
    const int lo   = lane & 15;
    const int hi   = lane >> 4;

    const int mtile = blockIdx.x * 8 + wave;
    if (mtile >= MTILES) return;           // wave-uniform; EXEC stays all-1s
    const int mBase = mtile * 16;

    const bf16_t* Arow = A + (size_t)(mBase + lo) * K;   // A: M = lo for all lanes
    const bf16_t* Bcol = sB + (size_t)lo * K;            // B: N-column = lo

    v8f acc = {0.f, 0.f, 0.f, 0.f, 0.f, 0.f, 0.f, 0.f};
#pragma unroll
    for (int k0 = 0; k0 < K; k0 += 32) {
        // A 16x32 bf16 layout: lanes0-15 K=0..7 & 16..23, lanes16-31 K=8..15 & 24..31
        bf16x8 a0 = *(const bf16x8*)(Arow + k0 + hi * 8);
        bf16x8 a1 = *(const bf16x8*)(Arow + k0 + 16 + hi * 8);
        bf16x16 af = __builtin_shufflevector(a0, a1,
            0,1,2,3,4,5,6,7,8,9,10,11,12,13,14,15);
        // B 32x16 bf16 layout: lanes0-15 K=0..7 & 16..23, lanes16-31 K=8..15 & 24..31
        bf16x8 b0 = *(const bf16x8*)(Bcol + k0 + hi * 8);
        bf16x8 b1 = *(const bf16x8*)(Bcol + k0 + 16 + hi * 8);
        bf16x16 bfrag = __builtin_shufflevector(b0, b1,
            0,1,2,3,4,5,6,7,8,9,10,11,12,13,14,15);
        acc = __builtin_amdgcn_wmma_f32_16x16x32_bf16(
            false, af, false, bfrag, (short)0, acc, false, false);
    }

    // C layout: VGPR r -> (M = mBase + hi*8 + r, N = nBase + lo)
    float* Cp = C + (size_t)(mBase + hi * 8) * N + nBase + lo;
#pragma unroll
    for (int r = 0; r < 8; ++r)
        Cp[(size_t)r * N] = acc[r];
}

// ---------------------------------------------------------------------------
// SpMM scatter: one wave per edge; lanes cover channels with float4 gathers
// and f32 atomic scatter. Prefetch next edge's source row.
// ---------------------------------------------------------------------------
__global__ __launch_bounds__(256) void spmm_atomic_kernel(
    const float* __restrict__ h, const float* __restrict__ vals,
    const int* __restrict__ rowi, const int* __restrict__ coli,
    float* __restrict__ agg, int C)
{
    const int lane = threadIdx.x & 31;
    const int wid  = blockIdx.x * (blockDim.x >> 5) + (threadIdx.x >> 5);
    const int nw   = gridDim.x * (blockDim.x >> 5);
    const int C4   = C >> 2;

    for (int e = wid; e < N_EDGES; e += nw) {
        const int en = e + nw;
        if (en < N_EDGES)
            __builtin_prefetch(h + (size_t)coli[en] * C, 0, 0);  // global_prefetch_b8

        const float v = vals[e];
        const float4* srcv = (const float4*)(h + (size_t)coli[e] * C);
        float* dst = agg + (size_t)rowi[e] * C;
        for (int j = lane; j < C4; j += 32) {
            float4 m = srcv[j];
            atomicAdd(dst + j * 4 + 0, v * m.x);
            atomicAdd(dst + j * 4 + 1, v * m.y);
            atomicAdd(dst + j * 4 + 2, v * m.z);
            atomicAdd(dst + j * 4 + 3, v * m.w);
        }
    }
}

// PReLU + deterministic dropout (keep where mask > 0.5, scale 2x), emit bf16
__global__ __launch_bounds__(256) void prelu_drop_bf16_kernel(
    const float* __restrict__ agg, const float* __restrict__ a,
    const float* __restrict__ mask, bf16_t* __restrict__ out, int total, int C)
{
    for (int i = blockIdx.x * blockDim.x + threadIdx.x; i < total;
         i += gridDim.x * blockDim.x) {
        float x = agg[i];
        float y = (x >= 0.f) ? x : a[i & (C - 1)] * x;
        y *= (mask[i] > 0.5f) ? 2.0f : 0.0f;
        out[i] = (bf16_t)y;
    }
}

// Final PReLU in place on d_out
__global__ __launch_bounds__(256) void prelu_inplace_kernel(
    float* __restrict__ out, const float* __restrict__ a, int total, int C)
{
    for (int i = blockIdx.x * blockDim.x + threadIdx.x; i < total;
         i += gridDim.x * blockDim.x) {
        float x = out[i];
        out[i] = (x >= 0.f) ? x : a[i & (C - 1)] * x;
    }
}

// ---------------------------------------------------------------------------
extern "C" void kernel_launch(void* const* d_in, const int* in_sizes, int n_in,
                              void* d_out, int out_size, void* d_ws, size_t ws_size,
                              hipStream_t stream) {
    const float* x    = (const float*)d_in[0];
    const float* vals = (const float*)d_in[1];
    const float* mask = (const float*)d_in[2];
    const float* W1   = (const float*)d_in[3];
    const float* b1   = (const float*)d_in[4];
    const float* a1   = (const float*)d_in[5];
    const float* W2   = (const float*)d_in[6];
    const float* b2   = (const float*)d_in[7];
    const float* a2   = (const float*)d_in[8];
    const int*   rowi = (const int*)d_in[9];
    const int*   coli = (const int*)d_in[10];
    float* out = (float*)d_out;

    // workspace layout (256-byte aligned offsets)
    char* ws = (char*)d_ws;
    bf16_t* xb   = (bf16_t*)(ws);                              // 12,800,000 B
    bf16_t* w1t  = (bf16_t*)(ws + 12800000);                   //     65,536 B  [NHID][NFEAT]
    bf16_t* w2t  = (bf16_t*)(ws + 12865536);                   //     65,536 B  [NOUT][NHID]
    float*  bufA = (float*) (ws + 12931072);                   // 51,200,000 B  h1 / later hb2(bf16)
    float*  bufB = (float*) (ws + 64131072);                   // 51,200,000 B  agg1 / later h2
    bf16_t* hb2  = (bf16_t*)bufA;                              // bf16 view reusing bufA

    // 1) convert inputs to bf16 (x) and transposed bf16 weight panels
    convert_bf16_kernel<<<2048, 256, 0, stream>>>(x, xb, N_NODES * NFEAT);
    transpose_bf16_kernel<<<128, 256, 0, stream>>>(W1, w1t, NFEAT, NHID);
    transpose_bf16_kernel<<<128, 256, 0, stream>>>(W2, w2t, NHID, NOUT);

    // 2) h1 = x @ W1   (bf16 WMMA, fp32 accumulate) -> bufA
    {
        dim3 grid((MTILES + 7) / 8, NHID / 16);
        gemm_bf16_wmma<NFEAT><<<grid, 256, 0, stream>>>(xb, w1t, bufA, NHID);
    }

    // 3) agg1 = A @ h1 + b1 -> bufB
    init_bias_kernel<<<2048, 256, 0, stream>>>(bufB, b1, N_NODES * NHID, NHID);
    spmm_atomic_kernel<<<2048, 256, 0, stream>>>(bufA, vals, rowi, coli, bufB, NHID);

    // 4) hb2 = bf16( dropout( prelu(agg1, a1) ) ) -> bufA (bf16 view)
    prelu_drop_bf16_kernel<<<2048, 256, 0, stream>>>(bufB, a1, mask, hb2,
                                                     N_NODES * NHID, NHID);

    // 5) h2 = hb2 @ W2 -> bufB (fp32)
    {
        dim3 grid((MTILES + 7) / 8, NOUT / 16);
        gemm_bf16_wmma<NHID><<<grid, 256, 0, stream>>>(hb2, w2t, bufB, NOUT);
    }

    // 6) out = prelu( A @ h2 + b2, a2 )
    init_bias_kernel<<<2048, 256, 0, stream>>>(out, b2, N_NODES * NOUT, NOUT);
    spmm_atomic_kernel<<<2048, 256, 0, stream>>>(bufB, vals, rowi, coli, out, NOUT);
    prelu_inplace_kernel<<<2048, 256, 0, stream>>>(out, a2, N_NODES * NOUT, NOUT);
}
